// NodeNet_25134148616720
// MI455X (gfx1250) — matrix operand order repaired and checked by reading, not compiled, not run on hardware
//
#include <hip/hip_runtime.h>
#include <hip/hip_bf16.h>

typedef __bf16 bf16_t;
typedef __attribute__((ext_vector_type(16))) __bf16 v16bf;
typedef __attribute__((ext_vector_type(8)))  __bf16 v8bf;
typedef __attribute__((ext_vector_type(8)))  float  v8f;

#define N_NODE_FEAT 128
#define D_IN 256
#define HIDDEN 256
#define D_OUT 128

// ---------------------------------------------------------------------------
// Zero-fill kernel (sums + counts), float4 wide.
// ---------------------------------------------------------------------------
__global__ void zero_kernel(float* __restrict__ p, long n) {
    long i = (long)blockIdx.x * blockDim.x + threadIdx.x;
    long base = i * 4;
    if (base + 3 < n) {
        *(float4*)(p + base) = make_float4(0.f, 0.f, 0.f, 0.f);
    } else {
        for (long j = base; j < n; ++j) p[j] = 0.f;
    }
}

// ---------------------------------------------------------------------------
// Pack f32 weights [K][N] row-major into WMMA bf16 B-fragment order:
//   idx = ((ktile*(N/16)+ntile)*32 + lane)*16 + e
//   lane<16 : col = ntile*16+lane,  K = ktile*32 + e       (e=0..15)
//   lane>=16: col = ntile*16+l-16,  K = ktile*32 + 16 + e
// => each lane's 16 elements are contiguous (one 32B load per fragment).
// ---------------------------------------------------------------------------
__global__ void pack_weights(const float* __restrict__ W, bf16_t* __restrict__ Wb,
                             int K, int N) {
    int idx = blockIdx.x * 256 + threadIdx.x;
    if (idx >= K * N) return;
    int e    = idx & 15;
    int lane = (idx >> 4) & 31;
    int f    = idx >> 9;
    int ntiles = N >> 4;
    int ktile = f / ntiles;
    int ntile = f - ktile * ntiles;
    int n = ntile * 16 + (lane & 15);
    int k = ktile * 32 + ((lane >> 4) << 4) + e;
    Wb[idx] = (__bf16)W[(size_t)k * N + n];
}

// ---------------------------------------------------------------------------
// Scatter-add: one wave per edge, lane covers 4 floats (32*4=128).
// HW f32 atomics land in L2 (sums = 51MB < 192MB L2).
// ---------------------------------------------------------------------------
__global__ void scatter_kernel(const float* __restrict__ edge_attr,
                               const int*   __restrict__ col,
                               float* __restrict__ sums,
                               float* __restrict__ counts,
                               int n_edges) {
    int e    = blockIdx.x * 8 + (threadIdx.x >> 5);
    int lane = threadIdx.x & 31;
    if (e >= n_edges) return;
    int c = col[e];
    float4 v = *(const float4*)(edge_attr + (size_t)e * 128 + lane * 4);
    float* dst = sums + (size_t)c * 128 + lane * 4;
    unsafeAtomicAdd(dst + 0, v.x);
    unsafeAtomicAdd(dst + 1, v.y);
    unsafeAtomicAdd(dst + 2, v.z);
    unsafeAtomicAdd(dst + 3, v.w);
    if (lane == 0) unsafeAtomicAdd(counts + c, 1.0f);
}

// ---------------------------------------------------------------------------
// WMMA helpers
// ---------------------------------------------------------------------------
__device__ __forceinline__ v8f wmma_bf16(v16bf a, v16bf b, v8f c) {
    return __builtin_amdgcn_wmma_f32_16x16x32_bf16(
        false, a, false, b, (short)0, c, false, false);
}

// Load all A fragments (2 strips x 8 k-tiles) for this wave's 32x256 LDS strip.
// A 16x32 order used here (contiguous k): lane<16 -> row=lane,   K in {kb..kb+15}, kb=0
//                        lane>=16-> row=lane-16, same with kb=16
__device__ __forceinline__ void load_A(const __bf16 (*act)[256], int lane,
                                       v16bf A[2][8]) {
    const int rb = lane & 15;
    const int kb = (lane >> 4) * 16;
#pragma unroll
    for (int s = 0; s < 2; ++s) {
        const int r = s * 16 + rb;
#pragma unroll
        for (int kt = 0; kt < 8; ++kt) {
            v8bf lo = *(const v8bf*)&act[r][kt * 32 + kb];
            v8bf hi = *(const v8bf*)&act[r][kt * 32 + kb + 8];
            A[s][kt] = __builtin_shufflevector(lo, hi,
                0,1,2,3,4,5,6,7,8,9,10,11,12,13,14,15);
        }
    }
}

// GEMM layer (K=256) with bias+ReLU, result written back (bf16) into the same
// LDS strip (A is already register-resident).
template <int NT>
__device__ __forceinline__ void layer_lds(const v16bf A[2][8],
                                          const bf16_t* __restrict__ Wb,
                                          const float* __restrict__ bias,
                                          int lane, __bf16 (*act)[256]) {
    const int nl   = lane & 15;
    const int mrow = (lane >> 4) * 8;   // C/D layout: lanes>=16 hold rows 8..15
#pragma unroll 1
    for (int nt = 0; nt < NT; ++nt) {
        float b = bias[nt * 16 + nl];
        v8f acc0 = {}; v8f acc1 = {};
#pragma unroll
        for (int kt = 0; kt < 8; ++kt) {
            v16bf B = *(const v16bf*)(Wb + (((size_t)(kt * NT + nt)) * 32 + lane) * 16);
            acc0 = wmma_bf16(A[0][kt], B, acc0);
            acc1 = wmma_bf16(A[1][kt], B, acc1);
        }
#pragma unroll
        for (int v = 0; v < 8; ++v) {
            float f0 = fmaxf(acc0[v] + b, 0.f);
            float f1 = fmaxf(acc1[v] + b, 0.f);
            act[mrow + v][nt * 16 + nl]      = (__bf16)f0;
            act[16 + mrow + v][nt * 16 + nl] = (__bf16)f1;
        }
    }
}

// Final layer: K=256, N=128, bias, f32 result straight to global (predicated).
__device__ __forceinline__ void layer_out(const v16bf A[2][8],
                                          const bf16_t* __restrict__ Wb,
                                          const float* __restrict__ bias,
                                          int lane, float* __restrict__ out,
                                          long row0, int n_nodes) {
    const int nl   = lane & 15;
    const int mrow = (lane >> 4) * 8;
#pragma unroll 1
    for (int nt = 0; nt < 8; ++nt) {
        float b = bias[nt * 16 + nl];
        v8f acc0 = {}; v8f acc1 = {};
#pragma unroll
        for (int kt = 0; kt < 8; ++kt) {
            v16bf B = *(const v16bf*)(Wb + (((size_t)(kt * 8 + nt)) * 32 + lane) * 16);
            acc0 = wmma_bf16(A[0][kt], B, acc0);
            acc1 = wmma_bf16(A[1][kt], B, acc1);
        }
#pragma unroll
        for (int v = 0; v < 8; ++v) {
            long r0 = row0 + mrow + v;
            if (r0 < n_nodes) out[r0 * 128 + nt * 16 + nl] = acc0[v] + b;
            long r1 = row0 + 16 + mrow + v;
            if (r1 < n_nodes) out[r1 * 128 + nt * 16 + nl] = acc1[v] + b;
        }
    }
}

// ---------------------------------------------------------------------------
// Fused MLP: 4 waves/block, each wave independently owns 32 rows through all
// three layers. 64KB LDS per block for activation staging.
// ---------------------------------------------------------------------------
__global__ __launch_bounds__(128)
void mlp_kernel(const float* __restrict__ x,
                const float* __restrict__ sums,
                const float* __restrict__ counts,
                const bf16_t* __restrict__ W1b, const float* __restrict__ b1,
                const bf16_t* __restrict__ W2b, const float* __restrict__ b2,
                const bf16_t* __restrict__ W3b, const float* __restrict__ b3,
                float* __restrict__ out, int n_nodes) {
    __shared__ __bf16 smem[4][32][256];   // 64 KB
    const int wave = threadIdx.x >> 5;
    const int lane = threadIdx.x & 31;
    const long row0 = (long)blockIdx.x * 128 + wave * 32;
    __bf16 (*act)[256] = smem[wave];

    // Stage concat(x, scatter-mean) as bf16. Lanes 0-15 cover cols 0-127 (x),
    // lanes 16-31 cover cols 128-255 (sums * 1/max(count,1)). Fully coalesced.
    const int c0 = lane * 8;
    for (int r = 0; r < 32; ++r) {
        long node = row0 + r;
        if (node >= n_nodes) node = n_nodes - 1;   // clamp (tail rows discarded)
        const float4* src;
        float scale;
        if (lane < 16) {
            src = (const float4*)(x + (size_t)node * 128 + c0);
            scale = 1.0f;
        } else {
            src = (const float4*)(sums + (size_t)node * 128 + (c0 - 128));
            scale = 1.0f / fmaxf(counts[node], 1.0f);
        }
        float4 a = src[0], c = src[1];
        v8bf o;
        o[0] = (__bf16)(a.x * scale); o[1] = (__bf16)(a.y * scale);
        o[2] = (__bf16)(a.z * scale); o[3] = (__bf16)(a.w * scale);
        o[4] = (__bf16)(c.x * scale); o[5] = (__bf16)(c.y * scale);
        o[6] = (__bf16)(c.z * scale); o[7] = (__bf16)(c.w * scale);
        *(v8bf*)&act[r][c0] = o;
    }

    v16bf A[2][8];                       // 128 VGPRs, register-resident per layer
    load_A(act, lane, A);
    layer_lds<16>(A, W1b, b1, lane, act);    // 256 -> 256, ReLU
    load_A(act, lane, A);
    layer_lds<16>(A, W2b, b2, lane, act);    // 256 -> 256, ReLU
    load_A(act, lane, A);
    layer_out(A, W3b, b3, lane, out, row0, n_nodes);  // 256 -> 128
}

// ---------------------------------------------------------------------------
extern "C" void kernel_launch(void* const* d_in, const int* in_sizes, int n_in,
                              void* d_out, int out_size, void* d_ws, size_t ws_size,
                              hipStream_t stream) {
    const float* x          = (const float*)d_in[0];
    const int*   edge_index = (const int*)  d_in[1];
    const float* edge_attr  = (const float*)d_in[2];
    const float* W1 = (const float*)d_in[3];
    const float* b1 = (const float*)d_in[4];
    const float* W2 = (const float*)d_in[5];
    const float* b2 = (const float*)d_in[6];
    const float* W3 = (const float*)d_in[7];
    const float* b3 = (const float*)d_in[8];
    float* out = (float*)d_out;

    const int NN = in_sizes[0] / N_NODE_FEAT;   // 100000
    const int NE = in_sizes[1] / 2;             // 1600000
    const int* col = edge_index + NE;           // edge_index[1]

    // Workspace layout
    char* ws = (char*)d_ws;
    size_t off = 0;
    float* sums   = (float*)(ws + off); off += (size_t)NN * 128 * sizeof(float);
    float* counts = (float*)(ws + off); off += (size_t)NN * sizeof(float);
    off = (off + 63) & ~(size_t)63;
    bf16_t* W1b = (bf16_t*)(ws + off); off += (size_t)256 * 256 * sizeof(bf16_t);
    bf16_t* W2b = (bf16_t*)(ws + off); off += (size_t)256 * 256 * sizeof(bf16_t);
    bf16_t* W3b = (bf16_t*)(ws + off); off += (size_t)256 * 128 * sizeof(bf16_t);

    // 1) zero sums+counts (contiguous)
    long zn = (long)NN * 128 + NN;
    long zblocks = (zn / 4 + 255) / 256 + 1;
    zero_kernel<<<(int)zblocks, 256, 0, stream>>>(sums, zn);

    // 2) pack weights into WMMA B-fragment order (bf16)
    pack_weights<<<(256 * 256 + 255) / 256, 256, 0, stream>>>(W1, W1b, 256, 256);
    pack_weights<<<(256 * 256 + 255) / 256, 256, 0, stream>>>(W2, W2b, 256, 256);
    pack_weights<<<(256 * 128 + 255) / 256, 256, 0, stream>>>(W3, W3b, 256, 128);

    // 3) scatter-add edge features + counts (HW f32 atomics into L2)
    scatter_kernel<<<(NE + 7) / 8, 256, 0, stream>>>(edge_attr, col, sums, counts, NE);

    // 4) fused 3-layer WMMA MLP
    mlp_kernel<<<(NN + 127) / 128, 128, 0, stream>>>(
        x, sums, counts, W1b, b1, W2b, b2, W3b, b3, out, NN);
}
